// GAU_19550691131610
// MI455X (gfx1250) — matrix operand order repaired and checked by reading, not compiled, not run on hardware
//
#include <hip/hip_runtime.h>
#include <hip/hip_bf16.h>
#include <math.h>

// ---------------------------------------------------------------------------
// GAU forward for MI455X (gfx1250, wave32, WMMA bf16 16x16x32).
// All five matmuls run through a single generic WMMA GEMM with templated
// epilogues. B operand always in (N x K) row-major layout for contiguous
// B-fragment loads. Global->LDS movement uses the CDNA5 async copy path
// (GLOBAL_LOAD_ASYNC_TO_LDS_B128 + s_wait_asynccnt) with double-buffered
// LDS, so the K-loop needs no data-staging VGPRs (fixes the scratch-spill
// loop seen with register staging).
// ---------------------------------------------------------------------------

typedef __bf16 bf16;
typedef __attribute__((ext_vector_type(16))) __bf16 bf16x16;
typedef __attribute__((ext_vector_type(8)))  __bf16 bf16x8;
typedef __attribute__((ext_vector_type(8)))  float  f32x8;

constexpr int BATCH = 4;
constexpr int S     = 4096;
constexpr int D     = 1024;
constexpr int QKD   = 128;
constexpr int H     = 2048;
constexpr int H2    = 4096;     // 2*H
constexpr int M_TOT = BATCH * S;
constexpr float LN_EPS = 1e-5f;

// GEMM tiling: 256 threads = 8 waves; 4 m-waves x 2 n-waves; wave tile 32x64.
constexpr int BM = 128;
constexpr int BN = 128;
constexpr int BK = 64;

__device__ __forceinline__ float silu_f(float v) {
    return v / (1.0f + __expf(-v));
}

// CDNA5 async copy: 16 bytes global -> LDS per lane, tracked by ASYNCcnt.
// LDS operand is the wave-relative byte offset (low 32 bits of the generic
// pointer to a __shared__ object).
__device__ __forceinline__ void async_copy_b128(bf16* lds_dst, const bf16* g_src) {
    unsigned int l = (unsigned int)(unsigned long long)lds_dst;
    asm volatile("global_load_async_to_lds_b128 %0, %1, off"
                 :: "v"(l), "v"(g_src)
                 : "memory");
}
__device__ __forceinline__ void wait_async0() {
    asm volatile("s_wait_asynccnt 0x0" ::: "memory");
}

// ---------------------------------------------------------------------------
// LayerNorm: one block per row of D=1024, 256 threads x 4 floats.
// Writes bf16 normed activations (input to both projection GEMMs).
// ---------------------------------------------------------------------------
__global__ __launch_bounds__(256) void gau_layernorm_kernel(
    const float* __restrict__ x, const float* __restrict__ g,
    const float* __restrict__ b, bf16* __restrict__ out)
{
    const int row = blockIdx.x;
    const size_t base = (size_t)row * D;
    const int t = threadIdx.x;

    float4 v = ((const float4*)(x + base))[t];
    float s  = v.x + v.y + v.z + v.w;
    float ss = v.x * v.x + v.y * v.y + v.z * v.z + v.w * v.w;

    #pragma unroll
    for (int o = 16; o > 0; o >>= 1) {
        s  += __shfl_xor(s,  o, 32);
        ss += __shfl_xor(ss, o, 32);
    }
    __shared__ float red[16];
    __shared__ float stats[2];
    const int wid = t >> 5;
    if ((t & 31) == 0) { red[wid] = s; red[8 + wid] = ss; }
    __syncthreads();
    if (t == 0) {
        float ts = 0.f, tss = 0.f;
        #pragma unroll
        for (int i = 0; i < 8; ++i) { ts += red[i]; tss += red[8 + i]; }
        float mu  = ts * (1.0f / D);
        float var = tss * (1.0f / D) - mu * mu;
        stats[0] = mu;
        stats[1] = rsqrtf(var + LN_EPS);
    }
    __syncthreads();
    const float mu = stats[0], r = stats[1];

    const int c = t * 4;
    out[base + c + 0] = (bf16)((v.x - mu) * r * g[c + 0] + b[c + 0]);
    out[base + c + 1] = (bf16)((v.y - mu) * r * g[c + 1] + b[c + 1]);
    out[base + c + 2] = (bf16)((v.z - mu) * r * g[c + 2] + b[c + 2]);
    out[base + c + 3] = (bf16)((v.w - mu) * r * g[c + 3] + b[c + 3]);
}

// ---------------------------------------------------------------------------
// Transpose fp32 (R x C) -> bf16 (C x R). One-time weight prep.
// ---------------------------------------------------------------------------
__global__ __launch_bounds__(256) void gau_transpose_f32_bf16_kernel(
    const float* __restrict__ src, bf16* __restrict__ dst, int R, int C)
{
    const int idx = blockIdx.x * 256 + threadIdx.x;
    if (idx >= R * C) return;
    const int r = idx / C;
    const int c = idx - r * C;
    dst[(size_t)c * R + r] = (bf16)src[idx];
}

// ---------------------------------------------------------------------------
// T5 relative bias depends only on (i - j): precompute 2S-1 diagonal values.
// diag[t] = rel_emb[bucket(i-j)] * sqrt(D), t = (i-j) + (S-1).
// ---------------------------------------------------------------------------
__global__ __launch_bounds__(256) void gau_relbias_kernel(
    const float* __restrict__ rel_emb, float* __restrict__ diag)
{
    const int t = blockIdx.x * 256 + threadIdx.x;
    if (t >= 2 * S - 1) return;
    const int n = t - (S - 1);          // n = i - j
    const int nb = 16, me = 8;          // NUM_BUCKETS/2, nb/2
    const int ret = (n < 0) ? nb : 0;
    const int na = (n < 0) ? -n : n;
    int bucket;
    if (na < me) {
        bucket = ret + na;
    } else {
        float lv = logf((float)na / (float)me) / logf(128.0f / (float)me) * (float)(nb - me);
        int vl = me + (int)lv;
        if (vl > nb - 1) vl = nb - 1;
        bucket = ret + vl;
    }
    diag[t] = rel_emb[bucket] * 32.0f;  // sqrt(1024) = 32
}

// ---------------------------------------------------------------------------
// Generic bf16 WMMA GEMM, C(M x N) = A(M x K, lda=K) * Bt(N x K, ldb=K)^T,
// batched via blockIdx.z. Templated epilogue.
// ---------------------------------------------------------------------------
struct GemmArgs {
    const bf16* A;
    const bf16* Bt;
    int M, N, K;
    long aBatch, bBatch;           // element strides per blockIdx.z
    // epilogue resources (unused ones are null)
    const float* nbias;            // per-column bias (bh / bqk / bo)
    const float* g0; const float* b0;   // os_gamma[0], os_beta[0]
    const float* g1; const float* b1;   // os_gamma[1], os_beta[1]
    const float* diag;             // rel-bias diagonal
    const bf16*  gate;             // (M_TOT x H)
    const float* resid;            // x (residual)
    bf16*  oA;                     // primary bf16 output
    bf16*  oB;                     // secondary bf16 output
    float* oF;                     // fp32 output (d_out)
};

// EPI 0: hidden = silu(.+bh); n<H -> vT(B,H,S), n>=H -> gate(M_TOT,H)
// EPI 1: g = silu(.+bqk); q=g*g0+b0 -> oA ; k=g*g1+b1 -> oB
// EPI 2: P = relu((.+diag[i-j])/S)^2 -> oA (B,S,S)
// EPI 3: Y = . * gate -> oA (M_TOT,H)
// EPI 4: out = . + bo + x -> oF
template <int EPI>
__global__ __launch_bounds__(256) void gau_gemm_kernel(GemmArgs args)
{
    __shared__ __align__(16) bf16 sA[2][BM * BK];   // 2 x 16 KB
    __shared__ __align__(16) bf16 sB[2][BN * BK];   // 2 x 16 KB

    const int z  = blockIdx.z;
    const bf16* A  = args.A  + (size_t)z * args.aBatch;
    const bf16* Bt = args.Bt + (size_t)z * args.bBatch;
    const int K  = args.K;
    const int m0 = blockIdx.y * BM;
    const int n0 = blockIdx.x * BN;

    const int t    = threadIdx.x;
    const int lane = t & 31;
    const int wid  = t >> 5;
    const int wm   = (wid & 3) * 32;   // wave m-offset in tile
    const int wn   = (wid >> 2) * 64;  // wave n-offset in tile
    const int half = lane >> 4;        // 0/1
    const int lcol = lane & 15;

    // cooperative tile load mapping: thread -> (row, 32-element half-row)
    const int lrow = t >> 1;
    const int lkc  = (t & 1) * 32;

    f32x8 acc[2][4];
    #pragma unroll
    for (int mt = 0; mt < 2; ++mt)
        #pragma unroll
        for (int nt = 0; nt < 4; ++nt)
            acc[mt][nt] = 0.0f;

    const bf16* gAr = A  + (size_t)(m0 + lrow) * K + lkc;   // row base, k=0
    const bf16* gBr = Bt + (size_t)(n0 + lrow) * K + lkc;

    const int ktiles = K / BK;

    // ---- async double-buffered pipeline ----
    // prologue: issue tile 0 into buffer 0 (4 x b128 each for A and B)
    {
        bf16* lA = &sA[0][lrow * BK + lkc];
        bf16* lB = &sB[0][lrow * BK + lkc];
        #pragma unroll
        for (int i = 0; i < 4; ++i) {
            async_copy_b128(lA + i * 8, gAr + i * 8);
            async_copy_b128(lB + i * 8, gBr + i * 8);
        }
    }

    for (int kt = 0; kt < ktiles; ++kt) {
        const int cur = kt & 1;

        wait_async0();       // this wave's async writes (tile kt) landed
        __syncthreads();     // all waves: tile kt visible; buffer 1-cur free

        if (kt + 1 < ktiles) {
            const bf16* gA = gAr + (size_t)(kt + 1) * BK;
            const bf16* gB = gBr + (size_t)(kt + 1) * BK;
            bf16* lA = &sA[1 - cur][lrow * BK + lkc];
            bf16* lB = &sB[1 - cur][lrow * BK + lkc];
            #pragma unroll
            for (int i = 0; i < 4; ++i) {
                async_copy_b128(lA + i * 8, gA + i * 8);
                async_copy_b128(lB + i * 8, gB + i * 8);
            }
        }

        #pragma unroll
        for (int s = 0; s < 2; ++s) {          // two k=32 WMMA steps per tile
            bf16x16 af[2], bf[4];
            #pragma unroll
            for (int mt = 0; mt < 2; ++mt) {
                const int mr = wm + mt * 16 + lcol;
                const int k0 = s * 32 + half * 16;
                af[mt] = *(const bf16x16*)&sA[cur][mr * BK + k0];
            }
            #pragma unroll
            for (int nt = 0; nt < 4; ++nt) {
                const int nr = wn + nt * 16 + lcol;
                const int k0 = s * 32 + half * 16;
                bf[nt] = *(const bf16x16*)&sB[cur][nr * BK + k0];
            }
            #pragma unroll
            for (int mt = 0; mt < 2; ++mt)
                #pragma unroll
                for (int nt = 0; nt < 4; ++nt)
                    acc[mt][nt] = __builtin_amdgcn_wmma_f32_16x16x32_bf16(
                        false, af[mt], false, bf[nt],
                        (short)0, acc[mt][nt], false, false);
        }
    }

    // ---- epilogue ----
    #pragma unroll
    for (int mt = 0; mt < 2; ++mt) {
        #pragma unroll
        for (int nt = 0; nt < 4; ++nt) {
            f32x8 a = acc[mt][nt];
            #pragma unroll
            for (int g = 0; g < 8; ++g) {
                const int em = m0 + wm + mt * 16 + g + half * 8;  // row in batch
                const int en = n0 + wn + nt * 16 + lcol;          // column
                const float val = a[g];
                if constexpr (EPI == 0) {
                    const float h = silu_f(val + args.nbias[en]);
                    if (en < H) {
                        const int bz = em >> 12;           // em / S
                        const int sr = em & (S - 1);
                        args.oA[((size_t)bz * H + en) * S + sr] = (bf16)h;  // vT
                    } else {
                        args.oB[(size_t)em * H + (en - H)] = (bf16)h;      // gate
                    }
                } else if constexpr (EPI == 1) {
                    const float h = silu_f(val + args.nbias[en]);
                    args.oA[(size_t)em * QKD + en] = (bf16)(h * args.g0[en] + args.b0[en]);
                    args.oB[(size_t)em * QKD + en] = (bf16)(h * args.g1[en] + args.b1[en]);
                } else if constexpr (EPI == 2) {
                    float tq = (val + args.diag[em - en + (S - 1)]) * (1.0f / S);
                    tq = fmaxf(tq, 0.0f);
                    args.oA[((size_t)z * S + em) * S + en] = (bf16)(tq * tq);
                } else if constexpr (EPI == 3) {
                    const size_t mg = (size_t)z * S + em;
                    args.oA[mg * H + en] = (bf16)(val * (float)args.gate[mg * H + en]);
                } else {  // EPI == 4
                    args.oF[(size_t)em * D + en] =
                        val + args.nbias[en] + args.resid[(size_t)em * D + en];
                }
            }
        }
    }
}

// ---------------------------------------------------------------------------
// Host-side orchestration
// ---------------------------------------------------------------------------
extern "C" void kernel_launch(void* const* d_in, const int* in_sizes, int n_in,
                              void* d_out, int out_size, void* d_ws, size_t ws_size,
                              hipStream_t stream)
{
    (void)in_sizes; (void)n_in; (void)out_size; (void)ws_size;
    const float* x       = (const float*)d_in[0];
    const float* ln_g    = (const float*)d_in[1];
    const float* ln_b    = (const float*)d_in[2];
    const float* Wh      = (const float*)d_in[3];
    const float* bh      = (const float*)d_in[4];
    const float* Wqk     = (const float*)d_in[5];
    const float* bqk     = (const float*)d_in[6];
    const float* os_gamma= (const float*)d_in[7];   // (2, QK)
    const float* os_beta = (const float*)d_in[8];   // (2, QK)
    const float* Wo      = (const float*)d_in[9];
    const float* bo      = (const float*)d_in[10];
    const float* rel_emb = (const float*)d_in[11];
    float* out = (float*)d_out;

    // workspace carve-out
    char* ws = (char*)d_ws;
    size_t off = 0;
    auto carve = [&](size_t bytes) {
        char* p = ws + off;
        off = (off + bytes + 255) & ~(size_t)255;
        return p;
    };
    bf16*  normed = (bf16*)carve((size_t)M_TOT * D * 2);
    bf16*  WhT    = (bf16*)carve((size_t)H2 * D * 2);
    bf16*  WqkT   = (bf16*)carve((size_t)QKD * D * 2);
    bf16*  WoT    = (bf16*)carve((size_t)D * H * 2);
    bf16*  vT     = (bf16*)carve((size_t)BATCH * H * S * 2);
    bf16*  gate   = (bf16*)carve((size_t)M_TOT * H * 2);
    bf16*  qb     = (bf16*)carve((size_t)M_TOT * QKD * 2);
    bf16*  kb     = (bf16*)carve((size_t)M_TOT * QKD * 2);
    bf16*  P      = (bf16*)carve((size_t)BATCH * S * S * 2);
    bf16*  Y      = (bf16*)carve((size_t)M_TOT * H * 2);
    float* diag   = (float*)carve((size_t)(2 * S - 1) * 4);

    // 1) weight prep (transpose to bf16, N x K layout)
    gau_transpose_f32_bf16_kernel<<<(D * H2) / 256, 256, 0, stream>>>(Wh, WhT, D, H2);
    gau_transpose_f32_bf16_kernel<<<(D * QKD) / 256, 256, 0, stream>>>(Wqk, WqkT, D, QKD);
    gau_transpose_f32_bf16_kernel<<<(H * D) / 256, 256, 0, stream>>>(Wo, WoT, H, D);
    gau_relbias_kernel<<<(2 * S - 1 + 255) / 256, 256, 0, stream>>>(rel_emb, diag);

    // 2) LayerNorm
    gau_layernorm_kernel<<<M_TOT, 256, 0, stream>>>(x, ln_g, ln_b, normed);

    GemmArgs ga{};

    // 3) hidden = silu(normed @ Wh + bh) -> vT (transposed) + gate
    ga = GemmArgs{};
    ga.A = normed; ga.Bt = WhT; ga.M = M_TOT; ga.N = H2; ga.K = D;
    ga.nbias = bh; ga.oA = vT; ga.oB = gate;
    gau_gemm_kernel<0><<<dim3(H2 / BN, M_TOT / BM, 1), 256, 0, stream>>>(ga);

    // 4) qk = silu(normed @ Wqk + bqk); q,k affine
    ga = GemmArgs{};
    ga.A = normed; ga.Bt = WqkT; ga.M = M_TOT; ga.N = QKD; ga.K = D;
    ga.nbias = bqk;
    ga.g0 = os_gamma; ga.b0 = os_beta;
    ga.g1 = os_gamma + QKD; ga.b1 = os_beta + QKD;
    ga.oA = qb; ga.oB = kb;
    gau_gemm_kernel<1><<<dim3(QKD / BN, M_TOT / BM, 1), 256, 0, stream>>>(ga);

    // 5) P = relu((q k^T + bias) / S)^2, per batch
    ga = GemmArgs{};
    ga.A = qb; ga.Bt = kb; ga.M = S; ga.N = S; ga.K = QKD;
    ga.aBatch = (long)S * QKD; ga.bBatch = (long)S * QKD;
    ga.diag = diag; ga.oA = P;
    gau_gemm_kernel<2><<<dim3(S / BN, S / BM, BATCH), 256, 0, stream>>>(ga);

    // 6) Y = (P @ v) * gate, per batch (v consumed transposed: (H x S))
    ga = GemmArgs{};
    ga.A = P; ga.Bt = vT; ga.M = S; ga.N = H; ga.K = S;
    ga.aBatch = (long)S * S; ga.bBatch = (long)H * S;
    ga.gate = gate; ga.oA = Y;
    gau_gemm_kernel<3><<<dim3(H / BN, S / BM, BATCH), 256, 0, stream>>>(ga);

    // 7) out = Y @ Wo + bo + x
    ga = GemmArgs{};
    ga.A = Y; ga.Bt = WoT; ga.M = M_TOT; ga.N = D; ga.K = H;
    ga.nbias = bo; ga.resid = x; ga.oF = out;
    gau_gemm_kernel<4><<<dim3(D / BN, M_TOT / BM, 1), 256, 0, stream>>>(ga);
}
